// Attentive_81518479278689
// MI455X (gfx1250) — compile-verified, matrix operand order checked by reading, not run on hardware
//
#include <hip/hip_runtime.h>
#include <hip/hip_bf16.h>

// ---------------------------------------------------------------------------
// Attentive captioner forward on MI455X (gfx1250, wave32).
// Dominant cost: final vocab projection 1280x32000x1024 (~84 GFLOP) ->
// v_wmma_f32_16x16x32_f16; W_mlp f16 (65MB) is L2-resident (192MB L2).
// Fast GEMM path (M%64==0, N%32==0, no guards): 64x32 tile per wave,
// 8 wmma : 6 fragment loads per 32-wide K step.  Guarded 32x32 path only
// for the tiny N=49 attention projections.
// ---------------------------------------------------------------------------

#define BB 64
#define TT 20
#define CC 2048
#define RR 49
#define HH 512
#define EE 256
#define VV 32000
#define KCAT (HH + EE + HH)   // 1280: [c_t | emb | h]

typedef __attribute__((ext_vector_type(8)))  _Float16 v8h;
typedef __attribute__((ext_vector_type(16))) _Float16 v16h;
typedef __attribute__((ext_vector_type(8)))  float    v8f;

// ---- WMMA fragment helpers -------------------------------------------------
// 16-bit A-matrix 16x32 layout (ISA 7.12.2): lane&15 selects row; half j of a
// lane holds K = (j/8)*16 + (lane/16)*8 + (j%8) -> two contiguous 8-half runs
// at k0 + (lane>>4)*8 and +16.  W[N,K] rows give the B fragment the same way.

__device__ __forceinline__ v16h ldfrag(const _Float16* __restrict__ p) {
  v8h lo = *(const v8h*)(p);
  v8h hi = *(const v8h*)(p + 16);
  v16h f;
#pragma unroll
  for (int i = 0; i < 8; ++i) { f[i] = lo[i]; f[i + 8] = hi[i]; }
  return f;
}

__device__ __forceinline__ v16h ldfrag_guard(const _Float16* __restrict__ mat,
                                             int row, int nrows, long ldk,
                                             int k0, unsigned lane) {
  v16h f = {};
  if (row < nrows) {
    const _Float16* p = mat + (long)row * ldk + k0 + ((lane >> 4) << 3);
    v8h lo = *(const v8h*)(p);
    v8h hi = *(const v8h*)(p + 16);
#pragma unroll
    for (int i = 0; i < 8; ++i) { f[i] = lo[i]; f[i + 8] = hi[i]; }
  }
  return f;
}

// C/D 16x16 f32 layout: lane&15 = column n, rows m = (lane>>4)*8 + v.
__device__ __forceinline__ void st_tile_fast(float* __restrict__ C, long ldc,
                                             int m0, int n0, v8f acc,
                                             const float* __restrict__ bias,
                                             int relu, unsigned lane) {
  int n = n0 + (int)(lane & 15u);
  float bv = bias ? bias[n] : 0.0f;
  float* p = C + (long)(m0 + (int)((lane >> 4) << 3)) * ldc + n;
#pragma unroll
  for (int v = 0; v < 8; ++v) {
    float x = acc[v] + bv;
    if (relu) x = fmaxf(x, 0.0f);
    p[(long)v * ldc] = x;
  }
}

__device__ __forceinline__ void st_tile_guard(float* __restrict__ C, long ldc,
                                              int m0, int M, int n0, int N,
                                              v8f acc, const float* __restrict__ bias,
                                              int relu, unsigned lane) {
  int n = n0 + (int)(lane & 15u);
  if (n >= N) return;
  float bv = bias ? bias[n] : 0.0f;
  int mbase = m0 + (int)((lane >> 4) << 3);
#pragma unroll
  for (int v = 0; v < 8; ++v) {
    int m = mbase + v;
    if (m < M) {
      float x = acc[v] + bv;
      if (relu) x = fmaxf(x, 0.0f);
      C[(long)m * ldc + n] = x;
    }
  }
}

// Fast path: C[M,N] = act(A[M,K] * W[N,K]^T + bias); M%64==0, N%32==0, K%32==0.
// One wave computes a 64x32 C tile: 4 A-frags x 2 B-frags -> 8 wmma / K-step.
__global__ __launch_bounds__(256) void wmma_gemm_nt_fast(
    const _Float16* __restrict__ A, const _Float16* __restrict__ W,
    float* __restrict__ C, int M, int N, int K, long ldc,
    const float* __restrict__ bias, int relu) {
  unsigned lane = threadIdx.x & 31u;
  unsigned wid  = blockIdx.x * (blockDim.x >> 5) + (threadIdx.x >> 5);
  int ntn = N >> 5;
  int ntm = M >> 6;
  if (wid >= (unsigned)(ntm * ntn)) return;           // wave-uniform exit
  int tm = (int)(wid / (unsigned)ntn);
  int tn = (int)(wid % (unsigned)ntn);
  int m0 = tm * 64, n0 = tn * 32;

  int r = (int)(lane & 15u);
  int kh = (int)((lane >> 4) << 3);
  const _Float16* pa = A + (long)(m0 + r) * K + kh;
  const _Float16* pb = W + (long)(n0 + r) * K + kh;
  long aRow16 = (long)K * 16;

  v8f a00 = {}, a01 = {}, a10 = {}, a11 = {};
  v8f a20 = {}, a21 = {}, a30 = {}, a31 = {};

  for (int k0 = 0; k0 < K; k0 += 32) {
    v16h fa0 = ldfrag(pa);
    v16h fa1 = ldfrag(pa + aRow16);
    v16h fa2 = ldfrag(pa + 2 * aRow16);
    v16h fa3 = ldfrag(pa + 3 * aRow16);
    v16h fb0 = ldfrag(pb);
    v16h fb1 = ldfrag(pb + aRow16);
    a00 = __builtin_amdgcn_wmma_f32_16x16x32_f16(false, fa0, false, fb0, (short)0, a00, false, false);
    a01 = __builtin_amdgcn_wmma_f32_16x16x32_f16(false, fa0, false, fb1, (short)0, a01, false, false);
    a10 = __builtin_amdgcn_wmma_f32_16x16x32_f16(false, fa1, false, fb0, (short)0, a10, false, false);
    a11 = __builtin_amdgcn_wmma_f32_16x16x32_f16(false, fa1, false, fb1, (short)0, a11, false, false);
    a20 = __builtin_amdgcn_wmma_f32_16x16x32_f16(false, fa2, false, fb0, (short)0, a20, false, false);
    a21 = __builtin_amdgcn_wmma_f32_16x16x32_f16(false, fa2, false, fb1, (short)0, a21, false, false);
    a30 = __builtin_amdgcn_wmma_f32_16x16x32_f16(false, fa3, false, fb0, (short)0, a30, false, false);
    a31 = __builtin_amdgcn_wmma_f32_16x16x32_f16(false, fa3, false, fb1, (short)0, a31, false, false);
    pa += 32;
    pb += 32;
  }
  st_tile_fast(C, ldc, m0,      n0,      a00, bias, relu, lane);
  st_tile_fast(C, ldc, m0,      n0 + 16, a01, bias, relu, lane);
  st_tile_fast(C, ldc, m0 + 16, n0,      a10, bias, relu, lane);
  st_tile_fast(C, ldc, m0 + 16, n0 + 16, a11, bias, relu, lane);
  st_tile_fast(C, ldc, m0 + 32, n0,      a20, bias, relu, lane);
  st_tile_fast(C, ldc, m0 + 32, n0 + 16, a21, bias, relu, lane);
  st_tile_fast(C, ldc, m0 + 48, n0,      a30, bias, relu, lane);
  st_tile_fast(C, ldc, m0 + 48, n0 + 16, a31, bias, relu, lane);
}

// Guarded path for ragged N (attention projections, N=49): 32x32 tile.
__global__ __launch_bounds__(256) void wmma_gemm_nt_guard(
    const _Float16* __restrict__ A, const _Float16* __restrict__ W,
    float* __restrict__ C, int M, int N, int K, long ldc,
    const float* __restrict__ bias, int relu) {
  unsigned lane = threadIdx.x & 31u;
  unsigned wid  = blockIdx.x * (blockDim.x >> 5) + (threadIdx.x >> 5);
  int ntn = (N + 31) >> 5;
  int ntm = (M + 31) >> 5;
  if (wid >= (unsigned)(ntm * ntn)) return;
  int tm = (int)(wid / (unsigned)ntn);
  int tn = (int)(wid % (unsigned)ntn);
  int m0 = tm * 32, n0 = tn * 32;

  v8f acc00 = {}, acc01 = {}, acc10 = {}, acc11 = {};
  int r = (int)(lane & 15u);
  for (int k0 = 0; k0 < K; k0 += 32) {
    v16h fa0 = ldfrag_guard(A, m0 + r,      M, K, k0, lane);
    v16h fa1 = ldfrag_guard(A, m0 + 16 + r, M, K, k0, lane);
    v16h fb0 = ldfrag_guard(W, n0 + r,      N, K, k0, lane);
    v16h fb1 = ldfrag_guard(W, n0 + 16 + r, N, K, k0, lane);
    acc00 = __builtin_amdgcn_wmma_f32_16x16x32_f16(false, fa0, false, fb0, (short)0, acc00, false, false);
    acc01 = __builtin_amdgcn_wmma_f32_16x16x32_f16(false, fa0, false, fb1, (short)0, acc01, false, false);
    acc10 = __builtin_amdgcn_wmma_f32_16x16x32_f16(false, fa1, false, fb0, (short)0, acc10, false, false);
    acc11 = __builtin_amdgcn_wmma_f32_16x16x32_f16(false, fa1, false, fb1, (short)0, acc11, false, false);
  }
  st_tile_guard(C, ldc, m0,      M, n0,      N, acc00, bias, relu, lane);
  st_tile_guard(C, ldc, m0,      M, n0 + 16, N, acc01, bias, relu, lane);
  st_tile_guard(C, ldc, m0 + 16, M, n0,      N, acc10, bias, relu, lane);
  st_tile_guard(C, ldc, m0 + 16, M, n0 + 16, N, acc11, bias, relu, lane);
}

// ---- staging / pointwise kernels ------------------------------------------

__global__ void k_f32_to_f16(const float* __restrict__ in, _Float16* __restrict__ out, long n) {
  long i = (long)blockIdx.x * blockDim.x + threadIdx.x;
  if (i < n) out[i] = (_Float16)in[i];
}

// feat [B,C,49] -> Aenc f16 [B*49, C] (transposed) + ag16 [B,C] (avg pool)
__global__ void k_encode_prep(const float* __restrict__ feat,
                              _Float16* __restrict__ Aenc,
                              _Float16* __restrict__ ag16) {
  int i = blockIdx.x * blockDim.x + threadIdx.x;     // b*C + c
  if (i >= BB * CC) return;
  int b = i / CC, c = i % CC;
  const float* p = feat + (long)i * RR;
  float s = 0.0f;
#pragma unroll 7
  for (int r = 0; r < RR; ++r) {
    float v = p[r];
    s += v;
    Aenc[((long)(b * RR + r)) * CC + c] = (_Float16)v;
  }
  ag16[i] = (_Float16)(s * (1.0f / (float)RR));
}

// Wcat [4H, H+E+H] = [W_ih | W_hh], bcat = b_ih + b_hh
__global__ void k_build_wcat(const float* __restrict__ Wih, const float* __restrict__ Whh,
                             const float* __restrict__ bih, const float* __restrict__ bhh,
                             _Float16* __restrict__ Wcat, float* __restrict__ bcat) {
  int i = blockIdx.x * blockDim.x + threadIdx.x;     // n*KCAT + k
  if (i >= 4 * HH * KCAT) return;
  int n = i / KCAT, k = i % KCAT;
  float v = (k < HH + EE) ? Wih[(long)n * (HH + EE) + k]
                          : Whh[(long)n * HH + (k - HH - EE)];
  Wcat[i] = (_Float16)v;
  if (k == 0) bcat[n] = bih[n] + bhh[n];
}

__global__ void k_init_state(const float* __restrict__ vg,
                             float* __restrict__ h, float* __restrict__ c) {
  int i = blockIdx.x * blockDim.x + threadIdx.x;
  if (i < BB * HH) { float v = vg[i]; h[i] = v; c[i] = v; }
}

// h f32 -> h16, and into x_cat[:, 768:1280]
__global__ void k_conv_h(const float* __restrict__ h, _Float16* __restrict__ h16,
                         _Float16* __restrict__ xcat) {
  int i = blockIdx.x * blockDim.x + threadIdx.x;     // b*H + j
  if (i >= BB * HH) return;
  _Float16 v = (_Float16)h[i];
  h16[i] = v;
  int b = i >> 9, j = i & (HH - 1);
  xcat[(long)b * KCAT + (HH + EE) + j] = v;
}

// x_cat[:, 512:768] = embed16[captions[:, t]]
__global__ void k_emb_fill(const int* __restrict__ cap, const _Float16* __restrict__ emb16,
                           _Float16* __restrict__ xcat, int t) {
  int b = blockIdx.x, e = threadIdx.x;               // 256 threads
  int w = cap[b * TT + t];
  xcat[(long)b * KCAT + HH + e] = emb16[(long)w * EE + e];
}

// z[b,r] = sum_k tanh(Vproj[b,r,k] + g[b,k]) * wh[k]
__global__ void k_attn_z(const float* __restrict__ Vproj, const float* __restrict__ g,
                         const float* __restrict__ wh, float* __restrict__ z) {
  int br = blockIdx.x;                                // b*49 + r
  int b = br / RR;
  int k = threadIdx.x;                                // 64 threads
  __shared__ float red[64];
  float v = 0.0f;
  if (k < RR) v = tanhf(Vproj[(long)br * RR + k] + g[b * RR + k]) * wh[k];
  red[k] = v;
  __syncthreads();
#pragma unroll
  for (int s = 32; s > 0; s >>= 1) {
    if (k < s) red[k] += red[k + s];
    __syncthreads();
  }
  if (k == 0) z[br] = red[0];
}

// softmax over r, c_t = sum_r alpha[r]*V[b,r,:]; write hiddens[:, t, 0:H] and
// x_cat[:, 0:H]
__global__ void k_attn_apply(const float* __restrict__ z, const float* __restrict__ V,
                             float* __restrict__ hiddens, _Float16* __restrict__ xcat,
                             int t) {
  int b = blockIdx.x;
  int tid = threadIdx.x;                              // 512 threads
  __shared__ float alpha[RR];
  __shared__ float sinv;
  if (tid == 0) {
    float m = -1e30f;
    for (int r = 0; r < RR; ++r) m = fmaxf(m, z[b * RR + r]);
    float s = 0.0f;
    for (int r = 0; r < RR; ++r) { float e = expf(z[b * RR + r] - m); alpha[r] = e; s += e; }
    sinv = 1.0f / s;
  }
  __syncthreads();
  float acc = 0.0f;
  int hh = tid;
#pragma unroll 7
  for (int r = 0; r < RR; ++r)
    acc += alpha[r] * V[((long)(b * RR + r)) * HH + hh];
  acc *= sinv;
  hiddens[((long)(b * TT + t)) * (2 * HH) + hh] = acc;
  xcat[(long)b * KCAT + hh] = (_Float16)acc;
}

// torch gate order (i, f, g, o)
__global__ void k_lstm_update(const float* __restrict__ gates, float* __restrict__ cst,
                              float* __restrict__ hst, float* __restrict__ hiddens, int t) {
  int b = blockIdx.x, j = threadIdx.x;                // 512 threads
  const float* g = gates + (long)b * 4 * HH;
  float gi = 1.0f / (1.0f + expf(-g[j]));
  float gf = 1.0f / (1.0f + expf(-g[HH + j]));
  float gg = tanhf(g[2 * HH + j]);
  float go = 1.0f / (1.0f + expf(-g[3 * HH + j]));
  long idx = (long)b * HH + j;
  float cn = gf * cst[idx] + gi * gg;
  float hn = go * tanhf(cn);
  cst[idx] = cn;
  hst[idx] = hn;
  hiddens[((long)(b * TT + t)) * (2 * HH) + HH + j] = hn;
}

// ---------------------------------------------------------------------------

static inline int cdiv(long a, long b) { return (int)((a + b - 1) / b); }

static inline void gemm(const _Float16* A, const _Float16* W, float* C,
                        int M, int N, int K, long ldc, const float* bias, int relu,
                        hipStream_t s) {
  if ((M & 63) == 0 && (N & 31) == 0) {
    int waves = (M >> 6) * (N >> 5);
    wmma_gemm_nt_fast<<<(waves + 7) / 8, 256, 0, s>>>(A, W, C, M, N, K, ldc, bias, relu);
  } else {
    int waves = ((M + 31) / 32) * ((N + 31) / 32);
    wmma_gemm_nt_guard<<<(waves + 7) / 8, 256, 0, s>>>(A, W, C, M, N, K, ldc, bias, relu);
  }
}

extern "C" void kernel_launch(void* const* d_in, const int* in_sizes, int n_in,
                              void* d_out, int out_size, void* d_ws, size_t ws_size,
                              hipStream_t stream) {
  const float* feat  = (const float*)d_in[0];
  const int*   cap   = (const int*)  d_in[1];
  // d_in[2] lengths: unused (reference runs full T)
  const float* W_a   = (const float*)d_in[3];
  const float* b_a   = (const float*)d_in[4];
  const float* W_b   = (const float*)d_in[5];
  const float* b_b   = (const float*)d_in[6];
  const float* embed = (const float*)d_in[7];
  const float* Wv    = (const float*)d_in[8];
  const float* Wg    = (const float*)d_in[9];
  const float* Wh    = (const float*)d_in[10];
  const float* W_ih  = (const float*)d_in[11];
  const float* W_hh  = (const float*)d_in[12];
  const float* b_ih  = (const float*)d_in[13];
  const float* b_hh  = (const float*)d_in[14];
  const float* W_mlp = (const float*)d_in[15];
  const float* b_mlp = (const float*)d_in[16];
  float* scores = (float*)d_out;

  char* ws = (char*)d_ws;
  size_t pos = 0;
  auto carve = [&](size_t bytes) -> void* {
    void* p = ws + pos;
    pos = (pos + bytes + 255) & ~(size_t)255;
    return p;
  };

  _Float16* Wa16   = (_Float16*)carve((size_t)HH * CC * 2);
  _Float16* Wb16   = (_Float16*)carve((size_t)HH * CC * 2);
  _Float16* Wv16   = (_Float16*)carve((size_t)RR * HH * 2);
  _Float16* Wg16   = (_Float16*)carve((size_t)RR * HH * 2);
  _Float16* Wcat16 = (_Float16*)carve((size_t)4 * HH * KCAT * 2);
  float*    bcat   = (float*)   carve((size_t)4 * HH * 4);
  _Float16* Wmlp16 = (_Float16*)carve((size_t)VV * 2 * HH * 2);
  _Float16* emb16  = (_Float16*)carve((size_t)VV * EE * 2);
  _Float16* Aenc   = (_Float16*)carve((size_t)BB * RR * CC * 2);
  _Float16* ag16   = (_Float16*)carve((size_t)BB * CC * 2);
  float*    Vf     = (float*)   carve((size_t)BB * RR * HH * 4);
  _Float16* V16    = (_Float16*)carve((size_t)BB * RR * HH * 2);
  float*    Vproj  = (float*)   carve((size_t)BB * RR * RR * 4);
  float*    vg     = (float*)   carve((size_t)BB * HH * 4);
  float*    hbuf   = (float*)   carve((size_t)BB * HH * 4);
  float*    cbuf   = (float*)   carve((size_t)BB * HH * 4);
  _Float16* h16    = (_Float16*)carve((size_t)BB * HH * 2);
  float*    gbuf   = (float*)   carve((size_t)BB * RR * 4);
  float*    zbuf   = (float*)   carve((size_t)BB * RR * 4);
  _Float16* xcat   = (_Float16*)carve((size_t)BB * KCAT * 2);
  float*    gates  = (float*)   carve((size_t)BB * 4 * HH * 4);
  float*    hid    = (float*)   carve((size_t)BB * TT * 2 * HH * 4);
  _Float16* hid16  = (_Float16*)carve((size_t)BB * TT * 2 * HH * 2);
  (void)ws_size; (void)n_in; (void)in_sizes; (void)out_size;

  // ---- one-time (per launch) weight staging to f16 ----
  k_f32_to_f16<<<cdiv((long)HH * CC, 256), 256, 0, stream>>>(W_a, Wa16, (long)HH * CC);
  k_f32_to_f16<<<cdiv((long)HH * CC, 256), 256, 0, stream>>>(W_b, Wb16, (long)HH * CC);
  k_f32_to_f16<<<cdiv((long)RR * HH, 256), 256, 0, stream>>>(Wv, Wv16, (long)RR * HH);
  k_f32_to_f16<<<cdiv((long)RR * HH, 256), 256, 0, stream>>>(Wg, Wg16, (long)RR * HH);
  k_f32_to_f16<<<cdiv((long)VV * 2 * HH, 256), 256, 0, stream>>>(W_mlp, Wmlp16, (long)VV * 2 * HH);
  k_f32_to_f16<<<cdiv((long)VV * EE, 256), 256, 0, stream>>>(embed, emb16, (long)VV * EE);
  k_build_wcat<<<cdiv((long)4 * HH * KCAT, 256), 256, 0, stream>>>(W_ih, W_hh, b_ih, b_hh, Wcat16, bcat);

  // ---- encoder ----
  k_encode_prep<<<cdiv((long)BB * CC, 256), 256, 0, stream>>>(feat, Aenc, ag16);
  // V = relu(Aenc[3136,2048] * W_a^T + b_a)
  gemm(Aenc, Wa16, Vf, BB * RR, HH, CC, HH, b_a, 1, stream);
  // v_g = relu(ag * W_b^T + b_b)
  gemm(ag16, Wb16, vg, BB, HH, CC, HH, b_b, 1, stream);
  k_f32_to_f16<<<cdiv((long)BB * RR * HH, 256), 256, 0, stream>>>(Vf, V16, (long)BB * RR * HH);
  // V_proj = V * Wv^T  -> [B*49, 49]
  gemm(V16, Wv16, Vproj, BB * RR, RR, HH, RR, nullptr, 0, stream);
  k_init_state<<<cdiv((long)BB * HH, 256), 256, 0, stream>>>(vg, hbuf, cbuf);

  // ---- LSTM with attention, T sequential steps ----
  for (int t = 0; t < TT; ++t) {
    k_conv_h<<<cdiv((long)BB * HH, 256), 256, 0, stream>>>(hbuf, h16, xcat);
    // g = h * Wg^T -> [B, 49]
    gemm(h16, Wg16, gbuf, BB, RR, HH, RR, nullptr, 0, stream);
    k_attn_z<<<BB * RR, 64, 0, stream>>>(Vproj, gbuf, Wh, zbuf);
    k_attn_apply<<<BB, HH, 0, stream>>>(zbuf, Vf, hid, xcat, t);
    k_emb_fill<<<BB, EE, 0, stream>>>(cap, emb16, xcat, t);
    // gates = [c_t|emb|h] * [W_ih|W_hh]^T + (b_ih+b_hh) -> [B, 4H]
    gemm(xcat, Wcat16, gates, BB, 4 * HH, KCAT, 4 * HH, bcat, 0, stream);
    k_lstm_update<<<BB, HH, 0, stream>>>(gates, cbuf, hbuf, hid, t);
  }

  // ---- final vocab projection: [1280,1024] x [32000,1024]^T ----
  k_f32_to_f16<<<cdiv((long)BB * TT * 2 * HH, 256), 256, 0, stream>>>(hid, hid16, (long)BB * TT * 2 * HH);
  gemm(hid16, Wmlp16, scores, BB * TT, VV, 2 * HH, VV, b_mlp, 0, stream);
}
